// Decoder_22720376996401
// MI455X (gfx1250) — matrix operand order repaired and checked by reading, not compiled, not run on hardware
//
#include <hip/hip_runtime.h>
#include <hip/hip_bf16.h>

// ---------------------------------------------------------------------------
// LSTM decoder w/ attention + big-vocab readout for MI455X (gfx1250, wave32).
//
//   P0  prep:    weights pre-packed into WMMA B-fragment layout (f16 pairs,
//                1KB contiguous per 16x32 tile -> b128 fragment loads),
//                K-major f16 for the VALU projections, embedding gather,
//                src_enc f16 convert, h/att state init.
//   P1  decoder: 16 persistent WGs, 2 grid barriers per step.
//                 - gate GEMM g = [emb|att|h] @ [W_ih;W_hh]^T (WMMA f16->f32)
//                 - per-WG (2 batch rows): LSTM pointwise (cell state in
//                   registers), masked softmax attention, att/dec projections.
//   P2  readout: logits = decs @ W_read^T in 128-wide vocab chunks; packed
//                B slabs staged into LDS via global_load_async_to_lds_b128
//                (ASYNCcnt + s_wait_asynccnt), WMMA accumulate, per-row
//                (max, argmax, sumexp) partials; pass2 merges chunks ->
//                logsumexp, loss, argmax.
// ---------------------------------------------------------------------------

#define T_     32
#define B_     32
#define S_     200
#define E_     512
#define H_     512
#define V_     50000
#define G4H    2048           // 4*H
#define NPAD   50048          // V padded to 391*128
#define NCH    391            // vocab chunks of 128
#define NTREAD (NPAD / 16)    // 3128 16-col tiles in W_read
#define ROWS   1024           // T*B
#define NWG_DEC 16

#if __has_builtin(__builtin_amdgcn_global_load_async_to_lds_b128) && \
    __has_builtin(__builtin_amdgcn_s_wait_asynccnt)
#define HAVE_ASYNC_LDS 1
#endif

typedef __attribute__((ext_vector_type(16))) _Float16 v16h;
typedef __attribute__((ext_vector_type(8)))  float    v8f;

// exact parameter type of the async-to-LDS builtins (from clang diagnostics):
// 'int __attribute__((vector_size(16)))' pointers in AS1 (global) / AS3 (LDS)
typedef int vi4 __attribute__((vector_size(16)));
typedef vi4 __attribute__((address_space(1)))* gvi4p;
typedef vi4 __attribute__((address_space(3)))* lvi4p;

__device__ __forceinline__ v8f zero8() {
  v8f z;
#pragma unroll
  for (int i = 0; i < 8; ++i) z[i] = 0.0f;
  return z;
}

// Assemble a 16-lane-striped fragment from two 16B register loads.
__device__ __forceinline__ v16h frag_from(uint4 q0, uint4 q1) {
  union { v16h v; uint4 q[2]; } f;
  f.q[0] = q0; f.q[1] = q1;
  return f.v;
}

__device__ __forceinline__ float wave_sum(float v) {
#pragma unroll
  for (int o = 16; o > 0; o >>= 1) v += __shfl_xor(v, o);
  return v;
}
__device__ __forceinline__ float wave_max(float v) {
#pragma unroll
  for (int o = 16; o > 0; o >>= 1) v = fmaxf(v, __shfl_xor(v, o));
  return v;
}
__device__ __forceinline__ float sigmoidf(float x) {
  return 1.0f / (1.0f + __expf(-x));
}

// Grid-wide barrier across NWG_DEC persistent workgroups (counter zeroed via
// hipMemsetAsync before each launch; monotonically increasing goal).
__device__ __forceinline__ void grid_barrier(unsigned* ctr, unsigned goal) {
  __syncthreads();
  if (threadIdx.x == 0) {
    __threadfence();
    atomicAdd(ctr, 1u);
    while (__hip_atomic_load(ctr, __ATOMIC_RELAXED, __HIP_MEMORY_SCOPE_AGENT) < goal)
      __builtin_amdgcn_s_sleep(2);
  }
  __syncthreads();
  __threadfence();
}

// ------------------------------ prep kernels -------------------------------

// Pack W (Nout x K row-major, f32) into WMMA-B fragment layout:
// dst dword d: i = d&7, l = (d>>3)&31, tile = d>>8, kc = tile/NT, nt = tile%NT
//   n = nt*16 + (l&15),  k = kc*32 + (i>>2)*16 + (l>>4)*8 + 2*(i&3)
//   dst[d] = pack(f16(W[n][k]), f16(W[n][k+1]))   (0 for n >= Nvalid padding)
__global__ void k_pack_b(const float* __restrict__ src, unsigned* __restrict__ dst,
                         int K, int NT, int Nvalid, int total) {
  for (int d = blockIdx.x * blockDim.x + threadIdx.x; d < total;
       d += gridDim.x * blockDim.x) {
    int i = d & 7;
    int l = (d >> 3) & 31;
    int tile = d >> 8;
    int kc = tile / NT, nt = tile % NT;
    int n = nt * 16 + (l & 15);
    int k = kc * 32 + ((i >> 2) << 4) + ((l >> 4) << 3) + 2 * (i & 3);
    unsigned lo = 0, hi = 0;
    if (n < Nvalid) {
      lo = __builtin_bit_cast(unsigned short, (_Float16)src[(size_t)n * K + k]);
      hi = __builtin_bit_cast(unsigned short, (_Float16)src[(size_t)n * K + k + 1]);
    }
    dst[d] = lo | (hi << 16);
  }
}

// dst[c*dld + r] = f16(src[r*cols + c])   (K-major f16 for VALU projections)
__global__ void k_transpose_f16(const float* __restrict__ src,
                                _Float16* __restrict__ dst,
                                int rows, int cols, int dld) {
  int n = rows * cols;
  for (int i = blockIdx.x * blockDim.x + threadIdx.x; i < n;
       i += gridDim.x * blockDim.x) {
    int r = i / cols, c = i % cols;
    dst[(size_t)c * dld + r] = (_Float16)src[i];
  }
}

__global__ void k_convert_f16(const float* __restrict__ src,
                              _Float16* __restrict__ dst, int n) {
  for (int i = blockIdx.x * blockDim.x + threadIdx.x; i < n;
       i += gridDim.x * blockDim.x)
    dst[i] = (_Float16)src[i];
}

__global__ void k_embed(const int* __restrict__ tgt_in,
                        const float* __restrict__ table,
                        _Float16* __restrict__ emb_h) {
  int n = T_ * B_ * E_;
  for (int i = blockIdx.x * blockDim.x + threadIdx.x; i < n;
       i += gridDim.x * blockDim.x) {
    int tb = i / E_, e = i % E_;
    emb_h[i] = (_Float16)table[(size_t)tgt_in[tb] * E_ + e];
  }
}

__global__ void k_init_state(const float* __restrict__ last_state,
                             _Float16* __restrict__ h_h,
                             _Float16* __restrict__ att_h) {
  int n = B_ * H_;
  for (int i = blockIdx.x * blockDim.x + threadIdx.x; i < n;
       i += gridDim.x * blockDim.x) {
    h_h[i]   = (_Float16)last_state[i];   // h0 = last_state[0]
    att_h[i] = (_Float16)0.0f;            // att_tm1 init = 0
  }
}

// ------------------------------ decoder ------------------------------------

__global__ __launch_bounds__(512, 1)
void k_decoder(const _Float16* __restrict__ emb_h,
               const unsigned* __restrict__ Wih_p,   // packed B, K=1024 N=2048
               const unsigned* __restrict__ Whh_p,   // packed B, K=512  N=2048
               const _Float16* __restrict__ Wattn_t, // 1024 x 512 K-major
               const _Float16* __restrict__ Wgen_t,  // 1536 x 512 K-major
               const float* __restrict__ b_ih, const float* __restrict__ b_hh,
               const float* __restrict__ b_gen,
               const _Float16* __restrict__ src_h,   // B x S x H f16
               const int* __restrict__ src_lens,
               const float* __restrict__ last_cell,
               _Float16* h_h, _Float16* att_h,
               float* g_ws,                          // 32 x 2048 gate scratch
               _Float16* decs_h,                     // T x B x H f16
               unsigned* ctr) {
  const int wg   = blockIdx.x;        // 0..15
  const int tid  = threadIdx.x;       // 0..511
  const int lane = tid & 31;
  const int wave = tid >> 5;          // 0..15
  const int b0   = wg * 2;            // this WG owns batch rows b0, b0+1

  __shared__ float h2[2][H_];
  __shared__ float ctx2[2][H_];
  __shared__ float at2[2][H_];
  __shared__ float emb2[2][H_];
  __shared__ float sc[2][208];

  // LSTM cell state lives in registers for the whole scan.
  float creg[2] = { last_cell[(size_t)b0 * H_ + tid],
                    last_cell[(size_t)(b0 + 1) * H_ + tid] };

  // gate-GEMM tile for this wave: 16x16 at (mrow, ntile*16)
  const int mrow  = (wave & 1) * 16;
  const int ntile = wg * 8 + (wave >> 1);          // 16-col tile index in 2048
  const int nb    = ntile * 16;

  // per-lane fragment bases (advance by pointer increments inside the loops)
  const int m     = lane & 15;
  const int koffA = (lane >> 4) << 3;              // halves
  const uint4* bih0 = (const uint4*)(Wih_p + (size_t)ntile * 256) + lane * 2;
  const uint4* bhh0 = (const uint4*)(Whh_p + (size_t)ntile * 256) + lane * 2;
  const int arow = mrow + m;

  unsigned goal = 0;
  for (int t = 0; t < T_; ++t) {
    // ---- Phase 1: g = [emb_t | att | h] @ [W_ih ; W_hh]^T (WMMA, N-split) --
    const _Float16* paE = emb_h + (size_t)t * B_ * E_ + (size_t)arow * E_ + koffA;
    const _Float16* paA = att_h + (size_t)arow * H_ + koffA;
    const _Float16* paH = h_h   + (size_t)arow * H_ + koffA;
    const uint4* bi = bih0;   // strides straight through kc = 0..31 of W_ih
    const uint4* bh = bhh0;
    v8f acc = zero8();
#pragma unroll 4
    for (int kc = 0; kc < 16; ++kc) {            // emb part (W_ih K 0..511)
      v16h a = frag_from(*(const uint4*)paE, *(const uint4*)(paE + 16));
      v16h b = frag_from(bi[0], bi[1]);
      acc = __builtin_amdgcn_wmma_f32_16x16x32_f16(false, a, false, b,
                                                   (short)0, acc, false, false);
      paE += 32; bi += 8192;                     // 128 tiles * 256 dw / 4
    }
#pragma unroll 4
    for (int kc = 0; kc < 16; ++kc) {            // att part (W_ih K 512..1023)
      v16h a = frag_from(*(const uint4*)paA, *(const uint4*)(paA + 16));
      v16h b = frag_from(bi[0], bi[1]);
      acc = __builtin_amdgcn_wmma_f32_16x16x32_f16(false, a, false, b,
                                                   (short)0, acc, false, false);
      paA += 32; bi += 8192;
    }
#pragma unroll 4
    for (int kc = 0; kc < 16; ++kc) {            // h part (W_hh)
      v16h a = frag_from(*(const uint4*)paH, *(const uint4*)(paH + 16));
      v16h b = frag_from(bh[0], bh[1]);
      acc = __builtin_amdgcn_wmma_f32_16x16x32_f16(false, a, false, b,
                                                   (short)0, acc, false, false);
      paH += 32; bh += 8192;
    }
    {  // store C tile (ISA 32-bit C layout: VGPR r -> rows r / r+8)
      const int n = nb + (lane & 15);
      const int rb = (lane >> 4) * 8;
#pragma unroll
      for (int r = 0; r < 8; ++r)
        g_ws[(size_t)(mrow + rb + r) * G4H + n] = acc[r];
    }
    goal += NWG_DEC; grid_barrier(ctr, goal);   // all of g visible

    // ---- Phase 2 (per-WG, 2 batch rows): LSTM pointwise + attention -------
#pragma unroll
    for (int j = 0; j < 2; ++j) {
      const int b = b0 + j;
      float gi = g_ws[(size_t)b * G4H + tid]        + b_ih[tid]        + b_hh[tid];
      float gf = g_ws[(size_t)b * G4H + 512 + tid]  + b_ih[512 + tid]  + b_hh[512 + tid];
      float gg = g_ws[(size_t)b * G4H + 1024 + tid] + b_ih[1024 + tid] + b_hh[1024 + tid];
      float go = g_ws[(size_t)b * G4H + 1536 + tid] + b_ih[1536 + tid] + b_hh[1536 + tid];
      float cn = sigmoidf(gf) * creg[j] + sigmoidf(gi) * tanhf(gg);
      float hn = sigmoidf(go) * tanhf(cn);
      creg[j] = cn;
      h2[j][tid] = hn;
      h_h[(size_t)b * H_ + tid] = (_Float16)hn;
      emb2[j][tid] = (float)emb_h[(size_t)t * B_ * E_ + (size_t)b * E_ + tid];
    }
    __syncthreads();

    // scores[b][s] = h . src_enc[b][s]  (wave per (j,s) pair, lanes over K)
    for (int p = wave; p < 2 * S_; p += 16) {
      const int j = p / S_, s = p % S_, b = b0 + j;
      const _Float16* sp = src_h + ((size_t)b * S_ + s) * H_;
      float a = 0.f;
#pragma unroll
      for (int i = 0; i < 16; ++i) {
        int k = lane + 32 * i;
        a += (float)sp[k] * h2[j][k];
      }
      a = wave_sum(a);
      if (lane == 0)
        sc[j][s] = (s < src_lens[b]) ? a : -__builtin_inff();
    }
    __syncthreads();

    // masked softmax over S (one wave per batch row)
    if (wave < 2) {
      const int j = wave;
      float mx = -__builtin_inff();
      for (int s = lane; s < S_; s += 32) mx = fmaxf(mx, sc[j][s]);
      mx = wave_max(mx);
      float sum = 0.f;
      for (int s = lane; s < S_; s += 32) {
        float e = __expf(sc[j][s] - mx);
        sc[j][s] = e;
        sum += e;
      }
      sum = wave_sum(sum);
      const float inv = 1.0f / sum;
      for (int s = lane; s < S_; s += 32) sc[j][s] *= inv;
    }
    __syncthreads();

    // ctx = alpha @ src_enc   (thread per hidden column)
#pragma unroll
    for (int j = 0; j < 2; ++j) {
      const int b = b0 + j;
      float a = 0.f;
      for (int s = 0; s < S_; ++s)
        a += sc[j][s] * (float)src_h[((size_t)b * S_ + s) * H_ + tid];
      ctx2[j][tid] = a;
    }
    __syncthreads();

    // att = tanh([ctx, h] @ W_attn^T)
#pragma unroll
    for (int j = 0; j < 2; ++j) {
      const int b = b0 + j;
      float a = 0.f;
      for (int k = 0; k < H_; ++k)
        a += (float)Wattn_t[(size_t)k * H_ + tid] * ctx2[j][k];
      for (int k = 0; k < H_; ++k)
        a += (float)Wattn_t[(size_t)(512 + k) * H_ + tid] * h2[j][k];
      float v = tanhf(a);
      at2[j][tid] = v;
      att_h[(size_t)b * H_ + tid] = (_Float16)v;
    }
    __syncthreads();

    // dec = tanh([emb, h, att] @ W_gen^T + b_gen)
#pragma unroll
    for (int j = 0; j < 2; ++j) {
      const int b = b0 + j;
      float a = b_gen[tid];
      for (int k = 0; k < H_; ++k)
        a += (float)Wgen_t[(size_t)k * H_ + tid] * emb2[j][k];
      for (int k = 0; k < H_; ++k)
        a += (float)Wgen_t[(size_t)(512 + k) * H_ + tid] * h2[j][k];
      for (int k = 0; k < H_; ++k)
        a += (float)Wgen_t[(size_t)(1024 + k) * H_ + tid] * at2[j][k];
      decs_h[((size_t)t * B_ + b) * H_ + tid] = (_Float16)tanhf(a);
    }
    goal += NWG_DEC; grid_barrier(ctr, goal);   // h/att visible for next step
  }
}

// ------------------------------ readout ------------------------------------

// Pass 1: one WG per (128 rows x 128 vocab cols) tile. Packed B slab (8KB,
// contiguous) staged into LDS via async-to-LDS (fallback: b128 copies);
// fragments via 2x ds_load_b128.
__global__ __launch_bounds__(256, 2)
void k_logits_pass1(const _Float16* __restrict__ decs_h,   // 1024 x 512
                    const unsigned* __restrict__ Wread_p,  // packed B layout
                    float* __restrict__ pmax, float* __restrict__ psum,
                    int* __restrict__ parg) {
  __shared__ __align__(16) unsigned lbs[2048];   // 8 tiles x 256 dwords
  const int tid   = threadIdx.x;
  const int lane  = tid & 31;
  const int wave  = tid >> 5;                 // 0..7
  const int nbase = blockIdx.y * 128;
  const int ntb   = blockIdx.y * 8;           // first 16-col tile of chunk
  const int mrow  = blockIdx.x * 128 + wave * 16;

  v8f acc[8];
#pragma unroll
  for (int c = 0; c < 8; ++c) acc[c] = zero8();

  const int m     = lane & 15;
  const int koffA = (lane >> 4) << 3;
  const _Float16* pa = decs_h + (size_t)(mrow + m) * H_ + koffA;
  const uint4* gslab = (const uint4*)(Wread_p + (size_t)ntb * 256);
  const size_t slab_stride = (size_t)NTREAD * 256 / 4;  // uint4 per k-chunk

  for (int kc = 0; kc < 16; ++kc) {
    const uint4* g4 = gslab;
    uint4* l4 = (uint4*)lbs;
#ifdef HAVE_ASYNC_LDS
    // gfx1250 async copy: 16B per lane, tracked by ASYNCcnt
    __builtin_amdgcn_global_load_async_to_lds_b128(
        (gvi4p)(g4 + tid), (lvi4p)(l4 + tid), 0, 0);
    __builtin_amdgcn_global_load_async_to_lds_b128(
        (gvi4p)(g4 + tid + 256), (lvi4p)(l4 + tid + 256), 0, 0);
    __builtin_amdgcn_s_wait_asynccnt(0);
#else
    l4[tid]       = g4[tid];
    l4[tid + 256] = g4[tid + 256];
#endif
    if (kc + 1 < 16)  // global_prefetch_b8 on next slab
      __builtin_prefetch((const void*)(gslab + slab_stride), 0, 1);
    __syncthreads();

    v16h a = frag_from(*(const uint4*)pa, *(const uint4*)(pa + 16));
    const uint4* lb = (const uint4*)lbs + lane * 2;
#pragma unroll
    for (int c = 0; c < 8; ++c) {
      v16h b = frag_from(lb[0], lb[1]);
      acc[c] = __builtin_amdgcn_wmma_f32_16x16x32_f16(false, a, false, b,
                                                      (short)0, acc[c],
                                                      false, false);
      lb += 64;                                  // next 16-col tile (256 dw)
    }
    __syncthreads();
    pa += 32;
    gslab += slab_stride;
  }

  // per-row chunk stats via half-wave (16-lane) butterflies
  const int half = lane >> 4;
  const int nl   = lane & 15;
#pragma unroll
  for (int r = 0; r < 8; ++r) {
    const int row = mrow + r + 8 * half;
    float vmax = -__builtin_inff();
    int   varg = 0x7fffffff;
#pragma unroll
    for (int c = 0; c < 8; ++c) {
      int n = nbase + c * 16 + nl;
      float v = acc[c][r];
      if (n < V_ && v > vmax) { vmax = v; varg = n; }
    }
#pragma unroll
    for (int o = 1; o <= 8; o <<= 1) {
      float om = __shfl_xor(vmax, o);
      int   oa = __shfl_xor(varg, o);
      if (om > vmax || (om == vmax && oa < varg)) { vmax = om; varg = oa; }
    }
    float s = 0.f;
#pragma unroll
    for (int c = 0; c < 8; ++c) {
      int n = nbase + c * 16 + nl;
      if (n < V_) s += __expf(acc[c][r] - vmax);
    }
#pragma unroll
    for (int o = 1; o <= 8; o <<= 1) s += __shfl_xor(s, o);
    if (nl == 0) {
      size_t idx = (size_t)row * NCH + blockIdx.y;
      pmax[idx] = vmax; psum[idx] = s; parg[idx] = varg;
    }
  }
}

// Pass 2: one wave per row — merge chunk partials, logsumexp, loss, argmax.
__global__ __launch_bounds__(256)
void k_logits_pass2(const float* __restrict__ pmax,
                    const float* __restrict__ psum,
                    const int* __restrict__ parg,
                    const _Float16* __restrict__ decs_h,
                    const float* __restrict__ W_read,   // V x H f32
                    const int* __restrict__ tgt_out,
                    float* __restrict__ out_loss, int* __restrict__ out_ys) {
  const int lane = threadIdx.x & 31;
  const int row  = blockIdx.x * 8 + (threadIdx.x >> 5);
  if (row >= ROWS) return;

  float vmax = -__builtin_inff();
  int   varg = 0x7fffffff;
  for (int j = lane; j < NCH; j += 32) {
    float m = pmax[(size_t)row * NCH + j];
    int   a = parg[(size_t)row * NCH + j];
    if (m > vmax || (m == vmax && a < varg)) { vmax = m; varg = a; }
  }
#pragma unroll
  for (int o = 1; o <= 16; o <<= 1) {
    float om = __shfl_xor(vmax, o);
    int   oa = __shfl_xor(varg, o);
    if (om > vmax || (om == vmax && oa < varg)) { vmax = om; varg = oa; }
  }
  float sum = 0.f;
  for (int j = lane; j < NCH; j += 32)
    sum += psum[(size_t)row * NCH + j] * __expf(pmax[(size_t)row * NCH + j] - vmax);
  sum = wave_sum(sum);
  const float lse = vmax + __logf(sum);

  const int tgt = tgt_out[row];
  float d = 0.f;
#pragma unroll
  for (int i = 0; i < 16; ++i) {
    int k = lane + 32 * i;
    d += (float)decs_h[(size_t)row * H_ + k] * W_read[(size_t)tgt * H_ + k];
  }
  d = wave_sum(d);

  if (lane == 0) {
    out_loss[row] = (tgt != 0) ? -(d - lse) : 0.0f;  // PAD == 0
    out_ys[row]   = varg;                            // argmax (int32)
  }
}

// ------------------------------ host side ----------------------------------

extern "C" void kernel_launch(void* const* d_in, const int* in_sizes, int n_in,
                              void* d_out, int out_size, void* d_ws, size_t ws_size,
                              hipStream_t stream) {
  (void)in_sizes; (void)n_in; (void)out_size; (void)ws_size;
  const int*   tgt_in    = (const int*)d_in[0];
  const int*   tgt_out   = (const int*)d_in[1];
  const int*   src_lens  = (const int*)d_in[2];
  const float* src_enc   = (const float*)d_in[3];
  const float* last_state= (const float*)d_in[4];
  const float* last_cell = (const float*)d_in[5];
  const float* embed_tab = (const float*)d_in[6];
  const float* W_ih      = (const float*)d_in[7];
  const float* W_hh      = (const float*)d_in[8];
  const float* b_ih      = (const float*)d_in[9];
  const float* b_hh      = (const float*)d_in[10];
  const float* W_attn    = (const float*)d_in[11];
  const float* W_gen     = (const float*)d_in[12];
  const float* b_gen     = (const float*)d_in[13];
  const float* W_read    = (const float*)d_in[14];

  // workspace carve-up (256B aligned)
  char* ws = (char*)d_ws;
  size_t off = 0;
  auto take = [&](size_t bytes) { char* p = ws + off; off = (off + bytes + 255) & ~(size_t)255; return p; };
  unsigned* ctr       = (unsigned*)take(256);
  unsigned* Wih_p     = (unsigned*)take((size_t)1024 * 2048 * 2);  // packed B
  unsigned* Whh_p     = (unsigned*)take((size_t)512  * 2048 * 2);  // packed B
  unsigned* Wread_p   = (unsigned*)take((size_t)512  * NPAD * 2);  // packed B
  _Float16* Wattn_t   = (_Float16*)take((size_t)1024 * 512  * 2);
  _Float16* Wgen_t    = (_Float16*)take((size_t)1536 * 512  * 2);
  _Float16* emb_h     = (_Float16*)take((size_t)T_ * B_ * E_ * 2);
  _Float16* src_h     = (_Float16*)take((size_t)B_ * S_ * H_ * 2);
  _Float16* h_h       = (_Float16*)take((size_t)B_ * H_ * 2);
  _Float16* att_h     = (_Float16*)take((size_t)B_ * H_ * 2);
  float*    g_ws      = (float*)take((size_t)B_ * G4H * 4);
  _Float16* decs_h    = (_Float16*)take((size_t)T_ * B_ * H_ * 2);
  float*    pmax      = (float*)take((size_t)ROWS * NCH * 4);
  float*    psum      = (float*)take((size_t)ROWS * NCH * 4);
  int*      parg      = (int*)take((size_t)ROWS * NCH * 4);

  float* out_loss = (float*)d_out;
  int*   out_ys   = (int*)d_out + ROWS;   // ys (int32) after losses

  // P0: prep — pack WMMA-B weights, K-major f16 projections, gathers
  k_pack_b<<<2048, 256, 0, stream>>>(W_ih,   Wih_p,   1024, 2048 / 16, 2048,
                                     1024 * 2048 / 2);
  k_pack_b<<<1024, 256, 0, stream>>>(W_hh,   Whh_p,   512,  2048 / 16, 2048,
                                     512 * 2048 / 2);
  k_pack_b<<<4096, 256, 0, stream>>>(W_read, Wread_p, 512,  NTREAD,    V_,
                                     512 * NPAD / 2);
  k_transpose_f16<<<512, 256, 0, stream>>>(W_attn, Wattn_t, 512, 1024, 512);
  k_transpose_f16<<<768, 256, 0, stream>>>(W_gen,  Wgen_t,  512, 1536, 512);
  k_convert_f16<<<4096, 256, 0, stream>>>(src_enc, src_h, B_ * S_ * H_);
  k_embed<<<1024, 256, 0, stream>>>(tgt_in, embed_tab, emb_h);
  k_init_state<<<64, 256, 0, stream>>>(last_state, h_h, att_h);

  // P1: persistent decoder (grid-barrier counter must start at 0 every call)
  (void)hipMemsetAsync(ctr, 0, 256, stream);
  k_decoder<<<NWG_DEC, 512, 0, stream>>>(emb_h, Wih_p, Whh_p, Wattn_t, Wgen_t,
                                         b_ih, b_hh, b_gen, src_h, src_lens,
                                         last_cell, h_h, att_h, g_ws, decs_h,
                                         ctr);

  // P2: readout
  k_logits_pass1<<<dim3(ROWS / 128, NCH), 256, 0, stream>>>(decs_h, Wread_p,
                                                            pmax, psum, parg);
  k_logits_pass2<<<ROWS / 8, 256, 0, stream>>>(pmax, psum, parg, decs_h,
                                               W_read, tgt_out,
                                               out_loss, out_ys);
}
